// MultiHopReasoner_66640712565462
// MI455X (gfx1250) — compile-verified, matrix operand order checked by reading, not run on hardware
//
#include <hip/hip_runtime.h>
#include <hip/hip_bf16.h>
#include <math.h>

#define NNODES 50000
#define NEDGES 800000
#define DIN    384
#define HID    256
#define HEADS  4
#define CDIM   64
#define EDIM   32
#define NLAYERS 2
#define E2 (NEDGES + NNODES)
#define SLOPE  0.2f
#define LN_EPS 1e-5f

typedef __attribute__((ext_vector_type(16))) __bf16 v16bf_t;
typedef __attribute__((ext_vector_type(8)))  float  v8f_t;

union AFrag { v16bf_t v; unsigned u[8]; };
union CFrag { v8f_t v; float f[8]; };

// fp32 -> bf16 round-to-nearest-even, as raw bits
__device__ __forceinline__ unsigned short f2bf(float f) {
  unsigned u = __float_as_uint(f);
  return (unsigned short)((u + 0x7FFFu + ((u >> 16) & 1u)) >> 16);
}
// pack two fp32 into one u32 of two bf16 (lo | hi<<16)
__device__ __forceinline__ unsigned pack_bf2(float lo, float hi) {
  return (unsigned)f2bf(lo) | ((unsigned)f2bf(hi) << 16);
}

// float atomic max via signed/unsigned integer ordering trick
__device__ __forceinline__ void atomicMaxF(float* addr, float val) {
  if (val >= 0.0f) atomicMax((int*)addr, __float_as_int(val));
  else             atomicMin((unsigned int*)addr, __float_as_uint(val));
}

// Build a 16x32 bf16 WMMA operand fragment from LDS stored as [row][16 kpairs]
// (u32 per bf16 pair). Layout per ISA: lane row = lane&15; lanes<16 hold
// K={0..7,16..23}, lanes>=16 hold K={8..15,24..31}; VGPR j packs consecutive K
// pairs -> contiguous u32 reads the compiler merges into ds_load_b128.
__device__ __forceinline__ v16bf_t frag16(const unsigned* tile, int lane) {
  const unsigned* p = tile + (lane & 15) * 16;
  const int kb2 = (lane & 16) ? 4 : 0;
  AFrag f;
#pragma unroll
  for (int j = 0; j < 4; ++j) {
    f.u[j]     = p[kb2 + j];      // K = kb + 2j, +1
    f.u[4 + j] = p[8 + kb2 + j];  // K = 16 + kb + 2j, +1
  }
  return f.v;
}

// ---------------------------------------------------------------------------
// C[M,N] = act(A[M,K] @ B[K,N] + bias).  Block tile 128x128, BK=32.
// 8 waves as 4x2; each wave computes a 32x64 tile = 2x4 WMMA bf16 accumulators
// (8 WMMAs per K-step from 6 fragments).  act: 0=none, 1=relu.
// Requires N % 128 == 0, K % 32 == 0 (true here: N=256, K in {384,256}).
// ---------------------------------------------------------------------------
__global__ __launch_bounds__(256)
void gemm_bf16_wmma(const float* __restrict__ A, const float* __restrict__ B,
                    const float* __restrict__ bias, float* __restrict__ C,
                    int M, int K, int N, int act) {
  __shared__ unsigned a_s[128 * 16];  // A tile: [m][kpair]  (8 KB)
  __shared__ unsigned b_s[128 * 16];  // B tile: [n][kpair]  (8 KB)
  const int tid  = threadIdx.x;
  const int lane = tid & 31;
  const int wave = tid >> 5;
  const int mbase = blockIdx.x * 128;
  const int nbase = blockIdx.y * 128;
  const int wm = (wave >> 1) * 32;   // wave row offset   (0,32,64,96)
  const int wn = (wave & 1) * 64;    // wave col offset   (0,64)

  CFrag c[2][4];
#pragma unroll
  for (int ti = 0; ti < 2; ++ti)
#pragma unroll
    for (int tj = 0; tj < 4; ++tj)
#pragma unroll
      for (int i = 0; i < 8; ++i) c[ti][tj].f[i] = 0.f;

  for (int k0 = 0; k0 < K; k0 += 32) {
    // ---- A: 128x32 floats = 1024 float4, 4 per thread; packed b64 stores ----
#pragma unroll
    for (int r = 0; r < 4; ++r) {
      int lin = tid + 256 * r;
      int row = lin >> 3;
      int kc  = (lin & 7) * 4;          // k offset of this float4
      int m   = mbase + row;
      float4 v = make_float4(0.f, 0.f, 0.f, 0.f);
      if (m < M) v = *(const float4*)(A + (size_t)m * K + k0 + kc);
      unsigned u0 = pack_bf2(v.x, v.y);
      unsigned u1 = pack_bf2(v.z, v.w);
      *(unsigned long long*)(a_s + row * 16 + (kc >> 1)) =
          ((unsigned long long)u1 << 32) | u0;
    }
    // ---- B: 32x128 floats; thread handles a (k,k+1) pair x 4 columns -------
    // slots: 16 kpairs x 32 col-groups = 512, 2 per thread
#pragma unroll
    for (int r = 0; r < 2; ++r) {
      int s    = tid + 256 * r;
      int pair = s >> 5;                // 0..15
      int n4   = (s & 31) * 4;          // 0..124
      const float* bp = B + (size_t)(k0 + 2 * pair) * N + nbase + n4;
      float4 lo = *(const float4*)bp;          // row k   = 2*pair
      float4 hi = *(const float4*)(bp + N);    // row k+1
      b_s[(n4 + 0) * 16 + pair] = pack_bf2(lo.x, hi.x);
      b_s[(n4 + 1) * 16 + pair] = pack_bf2(lo.y, hi.y);
      b_s[(n4 + 2) * 16 + pair] = pack_bf2(lo.z, hi.z);
      b_s[(n4 + 3) * 16 + pair] = pack_bf2(lo.w, hi.w);
    }
    __syncthreads();

    v16bf_t a0 = frag16(a_s + (wm +  0) * 16, lane);
    v16bf_t a1 = frag16(a_s + (wm + 16) * 16, lane);
    v16bf_t bf[4];
#pragma unroll
    for (int tj = 0; tj < 4; ++tj) bf[tj] = frag16(b_s + (wn + tj * 16) * 16, lane);
#pragma unroll
    for (int tj = 0; tj < 4; ++tj) {
      c[0][tj].v = __builtin_amdgcn_wmma_f32_16x16x32_bf16(false, a0, false, bf[tj],
                                                           (short)0, c[0][tj].v, false, false);
      c[1][tj].v = __builtin_amdgcn_wmma_f32_16x16x32_bf16(false, a1, false, bf[tj],
                                                           (short)0, c[1][tj].v, false, false);
    }
    __syncthreads();
  }

  // Epilogue. D layout: VGPR i, lane l -> (m = i + 8*(l>=16), n = l&15)
  const int mrow0 = mbase + wm + ((lane >> 4) << 3);
  const int ncol0 = nbase + wn + (lane & 15);
#pragma unroll
  for (int ti = 0; ti < 2; ++ti) {
#pragma unroll
    for (int tj = 0; tj < 4; ++tj) {
      int n = ncol0 + tj * 16;
      float bv = bias ? bias[n] : 0.f;
#pragma unroll
      for (int i = 0; i < 8; ++i) {
        int m = mrow0 + ti * 16 + i;
        if (m < M) {
          float v = c[ti][tj].f[i] + bv;
          if (act == 1) v = fmaxf(v, 0.f);
          C[(size_t)m * N + n] = v;
        }
      }
    }
  }
}

// ---------------------------------------------------------------------------
// Misc small kernels
// ---------------------------------------------------------------------------
__global__ void fill_kernel(float* p, float v, int n) {
  int t = blockIdx.x * blockDim.x + threadIdx.x;
  if (t < n) p[t] = v;
}

// deg[dst]++, loop_ea[dst][k] += edge_attr[e][k]
__global__ void degree_loopea_kernel(const int* __restrict__ ei, const float* __restrict__ ea,
                                     float* __restrict__ deg, float* __restrict__ loop_ea) {
  int t = blockIdx.x * blockDim.x + threadIdx.x;
  if (t >= NEDGES * EDIM) return;
  int e = t >> 5, k = t & 31;
  int d = ei[NEDGES + e];
  atomicAdd(&loop_ea[(size_t)d * EDIM + k], ea[t]);
  if (k == 0) atomicAdd(&deg[d], 1.0f);
}

__global__ void loopea_norm_kernel(float* __restrict__ loop_ea, const float* __restrict__ deg) {
  int t = blockIdx.x * blockDim.x + threadIdx.x;
  if (t >= NNODES * EDIM) return;
  loop_ea[t] /= fmaxf(deg[t >> 5], 1.0f);
}

// ---------------------------------------------------------------------------
// Edge message + attention logits. One wave per 16-edge group.
// ep = ea @ we computed on the fly: we in LDS, ea broadcast via readlane,
// we-column cached in 32 VGPRs and reused across the 16 edges.
// Lane l, channel set j -> ch = l + 32j, head = j>>1 (uniform per j).
// ---------------------------------------------------------------------------
__global__ __launch_bounds__(256)
void msg_logits_kernel(const float* __restrict__ xl, const float* __restrict__ xr,
                       const int* __restrict__ ei, const float* __restrict__ edge_attr,
                       const float* __restrict__ loop_ea, const float* __restrict__ we,
                       const float* __restrict__ att, float* __restrict__ logits,
                       float* __restrict__ lmax) {
  __shared__ float we_s[EDIM * HID];  // 32 KB
  __shared__ float att_s[HID];
  const int tid = threadIdx.x;
  for (int i = tid; i < EDIM * HID; i += 256) we_s[i] = we[i];
  if (tid < HID) att_s[tid] = att[tid];
  __syncthreads();

  const int lane  = tid & 31;
  const int wave  = tid >> 5;
  const int ebase = (blockIdx.x * 8 + wave) * 16;
  if (ebase >= E2) return;

  float eav[16]; int sreg[16], dreg[16];
#pragma unroll
  for (int it = 0; it < 16; ++it) {
    int e = ebase + it;
    int s = 0, d = 0; float ev = 0.f;
    if (e < E2) {
      if (e < NEDGES) {
        s = ei[e]; d = ei[NEDGES + e];
        ev = edge_attr[(size_t)e * EDIM + lane];
      } else {
        s = d = e - NEDGES;
        ev = loop_ea[(size_t)(e - NEDGES) * EDIM + lane];
      }
    }
    sreg[it] = s; dreg[it] = d; eav[it] = ev;
  }

#pragma unroll
  for (int h = 0; h < HEADS; ++h) {
    float acc[16];
#pragma unroll
    for (int it = 0; it < 16; ++it) acc[it] = 0.f;
#pragma unroll
    for (int jj = 0; jj < 2; ++jj) {
      const int ch = lane + 32 * (2 * h + jj);
      float wreg[32];
#pragma unroll
      for (int k = 0; k < 32; ++k) wreg[k] = we_s[k * HID + ch];
      const float attv = att_s[ch];
#pragma unroll
      for (int it = 0; it < 16; ++it) {
        float ep = 0.f;
#pragma unroll
        for (int k = 0; k < 32; ++k) ep = fmaf(__shfl(eav[it], k, 32), wreg[k], ep);
        float v = xl[(size_t)sreg[it] * HID + ch] + xr[(size_t)dreg[it] * HID + ch] + ep;
        v = (v > 0.f) ? v : SLOPE * v;          // leaky_relu
        acc[it] = fmaf(v, attv, acc[it]);
      }
    }
#pragma unroll
    for (int it = 0; it < 16; ++it) {
#pragma unroll
      for (int off = 16; off > 0; off >>= 1) acc[it] += __shfl_xor(acc[it], off, 32);
    }
    if (lane == 0) {
#pragma unroll
      for (int it = 0; it < 16; ++it) {
        int e = ebase + it;
        if (e < E2) {
          logits[(size_t)e * HEADS + h] = acc[it];
          atomicMaxF(&lmax[(size_t)dreg[it] * HEADS + h], acc[it]);
        }
      }
    }
  }
}

// ex = exp(logit - lmax[dst]); denom[dst] += ex
__global__ void softmax_ex_kernel(const float* __restrict__ logits, const float* __restrict__ lmax,
                                  const int* __restrict__ ei, float* __restrict__ ex,
                                  float* __restrict__ denom) {
  int t = blockIdx.x * blockDim.x + threadIdx.x;
  if (t >= E2 * HEADS) return;
  int e = t >> 2, h = t & 3;
  int d = (e < NEDGES) ? ei[NEDGES + e] : (e - NEDGES);
  float lm = lmax[(size_t)d * HEADS + h];
  if (lm < -1e37f) lm = 0.f;  // match jnp.where(isfinite, lmax, 0)
  float v = expf(logits[t] - lm);
  ex[t] = v;
  atomicAdd(&denom[(size_t)d * HEADS + h], v);
}

// agg[dst] += alpha * xl[src]; thread per (edge, 4 channels)
__global__ void aggregate_kernel(const float* __restrict__ xl, const int* __restrict__ ei,
                                 const float* __restrict__ ex, const float* __restrict__ denom,
                                 float* __restrict__ agg) {
  int t = blockIdx.x * blockDim.x + threadIdx.x;
  if (t >= E2 * 64) return;
  int e = t >> 6, q = t & 63;
  int h = q >> 4;
  int s, d;
  if (e < NEDGES) { s = ei[e]; d = ei[NEDGES + e]; }
  else            { s = d = e - NEDGES; }
  float alpha = ex[(size_t)e * HEADS + h] / (denom[(size_t)d * HEADS + h] + 1e-16f);
  float4 xv = *(const float4*)(xl + (size_t)s * HID + q * 4);
  float* o = agg + (size_t)d * HID + q * 4;
  atomicAdd(o + 0, alpha * xv.x);
  atomicAdd(o + 1, alpha * xv.y);
  atomicAdd(o + 2, alpha * xv.z);
  atomicAdd(o + 3, alpha * xv.w);
}

// h = h + elu(agg + b_conv); accumulate sum / sumsq for graph-layernorm
__global__ __launch_bounds__(256)
void post_stats_kernel(float* __restrict__ hbuf, const float* __restrict__ agg,
                       const float* __restrict__ bconv, float* __restrict__ stats) {
  __shared__ float s_sum[8], s_sq[8];
  int idx = blockIdx.x * 256 + threadIdx.x;
  float v = 0.f;
  if (idx < NNODES * HID) {
    int ch = idx & (HID - 1);
    float t = agg[idx] + bconv[ch];
    t = (t > 0.f) ? t : expm1f(t);   // elu
    v = hbuf[idx] + t;
    hbuf[idx] = v;
  }
  float s = v, q = v * v;
#pragma unroll
  for (int off = 16; off > 0; off >>= 1) {
    s += __shfl_xor(s, off, 32);
    q += __shfl_xor(q, off, 32);
  }
  int lane = threadIdx.x & 31, w = threadIdx.x >> 5;
  if (lane == 0) { s_sum[w] = s; s_sq[w] = q; }
  __syncthreads();
  if (threadIdx.x == 0) {
    float S = 0.f, Q = 0.f;
#pragma unroll
    for (int i = 0; i < 8; ++i) { S += s_sum[i]; Q += s_sq[i]; }
    atomicAdd(&stats[0], S);
    atomicAdd(&stats[1], Q);
  }
}

__global__ void layernorm_kernel(float* __restrict__ hbuf, const float* __restrict__ stats,
                                 const float* __restrict__ w, const float* __restrict__ b) {
  int idx = blockIdx.x * blockDim.x + threadIdx.x;
  if (idx >= NNODES * HID) return;
  const float Mf = (float)NNODES * (float)HID;
  float mu  = stats[0] / Mf;
  float var = stats[1] / Mf - mu * mu;
  float sd  = sqrtf(fmaxf(var, 0.f));
  int ch = idx & (HID - 1);
  hbuf[idx] = (hbuf[idx] - mu) / (sd + LN_EPS) * w[ch] + b[ch];
}

// ---------------------------------------------------------------------------
extern "C" void kernel_launch(void* const* d_in, const int* in_sizes, int n_in,
                              void* d_out, int out_size, void* d_ws, size_t ws_size,
                              hipStream_t stream) {
  (void)in_sizes; (void)n_in; (void)out_size; (void)ws_size;
  const float* x      = (const float*)d_in[0];
  const int*   ei     = (const int*)  d_in[1];
  const float* ea     = (const float*)d_in[2];
  const float* w_in   = (const float*)d_in[3];
  const float* b_in   = (const float*)d_in[4];
  const float* wl     = (const float*)d_in[5];
  const float* wr     = (const float*)d_in[6];
  const float* we     = (const float*)d_in[7];
  const float* att    = (const float*)d_in[8];
  const float* b_conv = (const float*)d_in[9];
  const float* ln_w   = (const float*)d_in[10];
  const float* ln_b   = (const float*)d_in[11];
  const float* w_out  = (const float*)d_in[12];
  const float* b_out  = (const float*)d_in[13];
  float* out = (float*)d_out;

  // workspace carve-up (~242 MB of floats)
  float* p = (float*)d_ws;
  float* deg     = p; p += NNODES;
  float* loop_ea = p; p += (size_t)NNODES * EDIM;
  float* xl      = p; p += (size_t)NNODES * HID;
  float* xr      = p; p += (size_t)NNODES * HID;
  float* agg     = p; p += (size_t)NNODES * HID;
  float* hbuf    = p; p += (size_t)NNODES * HID;
  float* logits  = p; p += (size_t)E2 * HEADS;
  float* exbuf   = p; p += (size_t)E2 * HEADS;
  float* lmax    = p; p += (size_t)NNODES * HEADS;
  float* denom   = p; p += (size_t)NNODES * HEADS;
  float* stats   = p; p += 2;

  auto cdiv = [](long a, long b) { return (int)((a + b - 1) / b); };
  const dim3 gemmBlk(256);
  const dim3 gemmGrid(cdiv(NNODES, 128), HID / 128);

  // --- self-loop edge_attr (fill_value='mean') ---
  fill_kernel<<<cdiv(NNODES, 256), 256, 0, stream>>>(deg, 0.f, NNODES);
  fill_kernel<<<cdiv((long)NNODES * EDIM, 256), 256, 0, stream>>>(loop_ea, 0.f, NNODES * EDIM);
  degree_loopea_kernel<<<cdiv((long)NEDGES * EDIM, 256), 256, 0, stream>>>(ei, ea, deg, loop_ea);
  loopea_norm_kernel<<<cdiv((long)NNODES * EDIM, 256), 256, 0, stream>>>(loop_ea, deg);

  // --- h = relu(x @ w_in + b_in) ---
  gemm_bf16_wmma<<<gemmGrid, gemmBlk, 0, stream>>>(x, w_in, b_in, hbuf, NNODES, DIN, HID, 1);

  for (int l = 0; l < NLAYERS; ++l) {
    // xl = h @ wl ; xr = h @ wr
    gemm_bf16_wmma<<<gemmGrid, gemmBlk, 0, stream>>>(
        hbuf, wl + (size_t)l * HID * HID, nullptr, xl, NNODES, HID, HID, 0);
    gemm_bf16_wmma<<<gemmGrid, gemmBlk, 0, stream>>>(
        hbuf, wr + (size_t)l * HID * HID, nullptr, xr, NNODES, HID, HID, 0);

    fill_kernel<<<cdiv((long)NNODES * HEADS, 256), 256, 0, stream>>>(lmax, -INFINITY, NNODES * HEADS);
    fill_kernel<<<cdiv((long)NNODES * HEADS, 256), 256, 0, stream>>>(denom, 0.f, NNODES * HEADS);
    fill_kernel<<<cdiv((long)NNODES * HID, 256), 256, 0, stream>>>(agg, 0.f, NNODES * HID);
    fill_kernel<<<1, 2, 0, stream>>>(stats, 0.f, 2);

    msg_logits_kernel<<<cdiv(E2, 128), 256, 0, stream>>>(
        xl, xr, ei, ea, loop_ea, we + (size_t)l * EDIM * HID,
        att + (size_t)l * HEADS * CDIM, logits, lmax);

    softmax_ex_kernel<<<cdiv((long)E2 * HEADS, 256), 256, 0, stream>>>(
        logits, lmax, ei, exbuf, denom);

    aggregate_kernel<<<cdiv((long)E2 * 64, 256), 256, 0, stream>>>(
        xl, ei, exbuf, denom, agg);

    post_stats_kernel<<<cdiv((long)NNODES * HID, 256), 256, 0, stream>>>(
        hbuf, agg, b_conv + (size_t)l * HID, stats);

    layernorm_kernel<<<cdiv((long)NNODES * HID, 256), 256, 0, stream>>>(
        hbuf, stats, ln_w + (size_t)l * HID, ln_b + (size_t)l * HID);
  }

  // --- out = h @ w_out + b_out ---
  gemm_bf16_wmma<<<gemmGrid, gemmBlk, 0, stream>>>(hbuf, w_out, b_out, out, NNODES, HID, HID, 0);
}
